// SelfAttentionWithGate_73967926771858
// MI455X (gfx1250) — compile-verified, matrix operand order checked by reading, not run on hardware
//
#include <hip/hip_runtime.h>
#include <hip/hip_bf16.h>

// Problem constants (reference: B=2, S=2048, d=1024, H=16, C=64)
#define B_  2
#define S_  2048
#define H_  16
#define C_  64
#define D_  1024   // C_QKV
#define HD_ 1024   // H_*C_

typedef __attribute__((ext_vector_type(16))) __bf16 v16bf;
typedef __attribute__((ext_vector_type(8)))  __bf16 v8bf;
typedef __attribute__((ext_vector_type(8)))  float  v8f;
typedef __attribute__((ext_vector_type(16))) float  v16f;

// 16-byte payload type for async global->LDS copies (matches builtin proto).
typedef int b128_t __attribute__((vector_size(16)));

#if __has_builtin(__builtin_amdgcn_global_load_async_to_lds_b128)
#define HAVE_ASYNC 1
#else
#define HAVE_ASYNC 0
#endif

#define AS1P(p) ((__attribute__((address_space(1))) b128_t*)(p))
#define AS3P(p) ((__attribute__((address_space(3))) b128_t*)(p))

// Native bf16 conversion (backend emits the hardware cvt op, RNE).
__device__ __forceinline__ unsigned short bfbits(float f) {
  __bf16 h = (__bf16)f;
  return __builtin_bit_cast(unsigned short, h);
}
__device__ __forceinline__ float bf2f(unsigned short s) {
  unsigned u = ((unsigned)s) << 16;
  return __builtin_bit_cast(float, u);
}

// Two 16B bf16 loads placed directly into the two halves of a WMMA operand.
__device__ __forceinline__ v16bf load16bf(const unsigned short* p0,
                                          const unsigned short* p1) {
  union { v16bf v; v8bf h[2]; } u;
  u.h[0] = *(const v8bf*)p0;
  u.h[1] = *(const v8bf*)p1;
  return u.v;
}

__device__ __forceinline__ v8f wmma_bf16(v16bf a, v16bf b, v8f c) {
  return __builtin_amdgcn_wmma_f32_16x16x32_bf16(false, a, false, b, (short)0, c,
                                                 false, false);
}

// ---------------------------------------------------------------------------
// Kernel 0: weight transpose+convert pre-pass.  Each 1024x1024 f32 weight is
// converted once to bf16 column-major (wT[N][K]) so GEMM B-tiles become two
// contiguous b128 loads.  gridDim.y = 5 selects wq/wk/wv/wg/wo.
// ---------------------------------------------------------------------------
__global__ __launch_bounds__(256) void wcvt_kernel(
    const float* __restrict__ wq, const float* __restrict__ wk,
    const float* __restrict__ wv, const float* __restrict__ wg,
    const float* __restrict__ wo, unsigned short* __restrict__ tq,
    unsigned short* __restrict__ tk, unsigned short* __restrict__ tv,
    unsigned short* __restrict__ tg, unsigned short* __restrict__ to_) {
  const int z = blockIdx.y;
  const float* W = (z == 0) ? wq : (z == 1) ? wk : (z == 2) ? wv
                 : (z == 3) ? wg : wo;
  unsigned short* T = (z == 0) ? tq : (z == 1) ? tk : (z == 2) ? tv
                    : (z == 3) ? tg : to_;
  const int n = blockIdx.x;        // output row (= weight column)
  const int k0 = threadIdx.x * 4;  // 256 threads x 4 = 1024 k values
  __bf16 h[4];
#pragma unroll
  for (int j = 0; j < 4; ++j)
    h[j] = (__bf16)W[(size_t)(k0 + j) * D_ + n];
  __builtin_memcpy(&T[(size_t)n * D_ + k0], h, 8);
}

// ---------------------------------------------------------------------------
// Kernel 1: fused QKV+Gate projections from pre-transposed bf16 weights.
// gridDim.z = 4 selects q/k/v/g.  Each wave computes a 16x64 strip
// (4 WMMA accumulators, A-tile reused 4x).
// Epilogues: z=0 q*1/sqrt(C) -> [B,H,S,C]; z=1 k -> [B,H,S,C];
//            z=2 v -> transposed [B,H,C,S]; z=3 sigmoid(g+bg) -> [B,H,S,C].
// ---------------------------------------------------------------------------
__global__ __launch_bounds__(256) void proj_kernel(
    const float* __restrict__ x, const unsigned short* __restrict__ tq,
    const unsigned short* __restrict__ tk, const unsigned short* __restrict__ tv,
    const unsigned short* __restrict__ tg, const float* __restrict__ bg,
    unsigned short* __restrict__ qws, unsigned short* __restrict__ kws,
    unsigned short* __restrict__ vtws, unsigned short* __restrict__ gws) {
  const int lane = threadIdx.x & 31;
  const int wave = threadIdx.x >> 5;
  const int ln = lane & 15;    // A: row m; B/C: column n
  const int half = lane >> 4;  // selects K sub-range / C row group
  const int t = blockIdx.x * 8 + wave;  // strip id over 256 x 16
  const int strip = t & (HD_ / 64 - 1); // 16 column strips of width 64
  const int tm = t >> 4;
  const int row0 = tm * 16, col0 = strip * 64;
  const int z = blockIdx.z;
  const unsigned short* WT = (z == 0) ? tq : (z == 1) ? tk : (z == 2) ? tv : tg;

  v8f acc[4] = {{}, {}, {}, {}};
  for (int k0 = 0; k0 < D_; k0 += 32) {
    // A tile: x[row0+ln][k0 + kmap(e,half)], two contiguous groups of 8 floats
    const float* ap = x + (size_t)(row0 + ln) * D_ + k0 + half * 8;
    union { v16f v; float4 q[4]; } ua;
    ua.q[0] = *(const float4*)(ap);
    ua.q[1] = *(const float4*)(ap + 4);
    ua.q[2] = *(const float4*)(ap + 16);
    ua.q[3] = *(const float4*)(ap + 20);
    v16bf A = __builtin_convertvector(ua.v, v16bf);
#pragma unroll
    for (int ct = 0; ct < 4; ++ct) {
      const unsigned short* wrow =
          WT + (size_t)(col0 + ct * 16 + ln) * D_ + k0;
      v16bf Bt = load16bf(wrow + half * 8, wrow + 16 + half * 8);
      acc[ct] = wmma_bf16(A, Bt, acc[ct]);
    }
  }

  // Epilogue: branch hoisted outside the unrolled store loops.
  const int grow0 = row0 + half * 8;       // rows grow0..grow0+7
  const int b = grow0 >> 11;               // constant within a tile row group
  const int s0 = grow0 & (S_ - 1);
  if (z == 0) {
#pragma unroll
    for (int ct = 0; ct < 4; ++ct) {
      const int gcol = col0 + ct * 16 + ln;
      const size_t base = ((size_t)(b * H_ + (gcol >> 6)) * S_) * C_ + (gcol & 63);
#pragma unroll
      for (int r = 0; r < 8; ++r)
        qws[base + (size_t)(s0 + r) * C_] = bfbits(acc[ct][r] * 0.125f);
    }
  } else if (z == 1) {
#pragma unroll
    for (int ct = 0; ct < 4; ++ct) {
      const int gcol = col0 + ct * 16 + ln;
      const size_t base = ((size_t)(b * H_ + (gcol >> 6)) * S_) * C_ + (gcol & 63);
#pragma unroll
      for (int r = 0; r < 8; ++r)
        kws[base + (size_t)(s0 + r) * C_] = bfbits(acc[ct][r]);
    }
  } else if (z == 2) {
#pragma unroll
    for (int ct = 0; ct < 4; ++ct) {
      const int gcol = col0 + ct * 16 + ln;
      const size_t base =
          ((size_t)(b * H_ + (gcol >> 6)) * C_ + (gcol & 63)) * S_ + s0;
#pragma unroll
      for (int r = 0; r < 8; ++r)
        vtws[base + r] = bfbits(acc[ct][r]);   // V transposed [B,H,C,S]
    }
  } else {
#pragma unroll
    for (int ct = 0; ct < 4; ++ct) {
      const int gcol = col0 + ct * 16 + ln;
      const float bgv = bg[gcol];
      const size_t base = ((size_t)(b * H_ + (gcol >> 6)) * S_) * C_ + (gcol & 63);
#pragma unroll
      for (int r = 0; r < 8; ++r) {
        const float g = 1.0f / (1.0f + __expf(-(acc[ct][r] + bgv)));
        gws[base + (size_t)(s0 + r) * C_] = bfbits(g);
      }
    }
  }
}

// ---------------------------------------------------------------------------
// Kernel 2: flash-attention per (b, h, 16-query tile) wave.
// Bias (the dominant 268 MB HBM stream) is double-buffered into LDS with
// GLOBAL_LOAD_ASYNC_TO_LDS_B128 + s_wait_asynccnt when available.
// Online softmax in f32; P transposed via LDS; gate fused into epilogue.
// ---------------------------------------------------------------------------
__global__ __launch_bounds__(128) void attn_kernel(
    const unsigned short* __restrict__ qws, const unsigned short* __restrict__ kws,
    const unsigned short* __restrict__ vtws, const unsigned short* __restrict__ gws,
    const float* __restrict__ mask, const float* __restrict__ bias,
    unsigned short* __restrict__ aout) {
  __shared__ __align__(16) unsigned short ptile[4][16 * 32];
#if HAVE_ASYNC
  __shared__ __align__(16) float btile[4][2][16 * 32];
#endif

  const int lane = threadIdx.x & 31;
  const int wave = threadIdx.x >> 5;
  const int ln = lane & 15;
  const int half = lane >> 4;
  const int h = blockIdx.y, b = blockIdx.z;
  const int q0 = (blockIdx.x * 4 + wave) * 16;

  const size_t bhS = (size_t)(b * H_ + h) * S_;
  const size_t bhC = (size_t)(b * H_ + h) * C_;

  // Q tile 16x64 in A layout: qa0 covers c=0..31, qa1 covers c=32..63
  const unsigned short* qrow = qws + (bhS + q0 + ln) * C_;
  v16bf qa0 = load16bf(qrow + half * 8, qrow + 16 + half * 8);
  v16bf qa1 = load16bf(qrow + 32 + half * 8, qrow + 48 + half * 8);

#if HAVE_ASYNC
  // Async-copy one 16x32 f32 bias tile (2 KB) into LDS: 4 x b128 per wave.
  auto issue_bias = [&](int kbn, int buf) {
#pragma unroll
    for (int j = 0; j < 4; ++j) {
      const int chunk = j * 32 + lane;
      const int row = chunk >> 3;
      const int c4 = (chunk & 7) * 4;
      const float* g = bias + ((size_t)h * S_ + q0 + row) * S_ + kbn + c4;
      __builtin_amdgcn_global_load_async_to_lds_b128(
          AS1P(g), AS3P(&btile[wave][buf][row * 32 + c4]), 0, 0);
    }
  };
  issue_bias(0, 0);
#endif

  float mrow[8], lrow[8];
  v8f oacc[4] = {{}, {}, {}, {}};
#pragma unroll
  for (int r = 0; r < 8; ++r) { mrow[r] = -1e30f; lrow[r] = 0.0f; }

  for (int kb = 0; kb < S_; kb += 32) {
#if HAVE_ASYNC
    const int buf = (kb >> 5) & 1;
    asm volatile("s_wait_asynccnt 0x0" ::: "memory");
    if (kb + 32 < S_) issue_bias(kb + 32, buf ^ 1);
#endif
    float sv[2][8];
#pragma unroll
    for (int kt = 0; kt < 2; ++kt) {
      const int key = kb + kt * 16 + ln;
      const unsigned short* krow = kws + (bhS + key) * C_;
      v16bf kb0 = load16bf(krow + half * 8, krow + 16 + half * 8);
      v16bf kb1 = load16bf(krow + 32 + half * 8, krow + 48 + half * 8);
      v8f sc = {};
      sc = wmma_bf16(qa0, kb0, sc);
      sc = wmma_bf16(qa1, kb1, sc);
      const float mterm = (mask[(size_t)b * S_ + key] - 1.0f) * 1e9f;
#pragma unroll
      for (int r = 0; r < 8; ++r) {
#if HAVE_ASYNC
        const float bterm = btile[wave][buf][(half * 8 + r) * 32 + kt * 16 + ln];
#else
        const int qr = q0 + half * 8 + r;
        const float bterm = bias[((size_t)h * S_ + qr) * S_ + key];
#endif
        sv[kt][r] = sc[r] + bterm + mterm;
      }
    }

    // per-row block max across both 16-key tiles and 16 lanes of the half
    float mx[8];
#pragma unroll
    for (int r = 0; r < 8; ++r) mx[r] = fmaxf(sv[0][r], sv[1][r]);
#pragma unroll
    for (int d = 1; d < 16; d <<= 1)
#pragma unroll
      for (int r = 0; r < 8; ++r) mx[r] = fmaxf(mx[r], __shfl_xor(mx[r], d, 32));

    float scl[8];
#pragma unroll
    for (int r = 0; r < 8; ++r) {
      const float mn = fmaxf(mrow[r], mx[r]);
      scl[r] = __expf(mrow[r] - mn);
      mrow[r] = mn;
      sv[0][r] = __expf(sv[0][r] - mn);
      sv[1][r] = __expf(sv[1][r] - mn);
    }
    float ss[8];
#pragma unroll
    for (int r = 0; r < 8; ++r) ss[r] = sv[0][r] + sv[1][r];
#pragma unroll
    for (int d = 1; d < 16; d <<= 1)
#pragma unroll
      for (int r = 0; r < 8; ++r) ss[r] += __shfl_xor(ss[r], d, 32);
#pragma unroll
    for (int r = 0; r < 8; ++r) {
      lrow[r] = lrow[r] * scl[r] + ss[r];
      oacc[0][r] *= scl[r]; oacc[1][r] *= scl[r];
      oacc[2][r] *= scl[r]; oacc[3][r] *= scl[r];
    }

    // P (C layout) -> LDS -> A layout
#pragma unroll
    for (int kt = 0; kt < 2; ++kt)
#pragma unroll
      for (int r = 0; r < 8; ++r)
        ptile[wave][(half * 8 + r) * 32 + kt * 16 + ln] = bfbits(sv[kt][r]);
    asm volatile("s_wait_dscnt 0x0" ::: "memory");
    const unsigned short* prow = &ptile[wave][ln * 32];
    v16bf pA = load16bf(prow + half * 8, prow + 16 + half * 8);

    // PV: 4 output column tiles, V^T rows contiguous over keys
#pragma unroll
    for (int ct = 0; ct < 4; ++ct) {
      const unsigned short* vrow = vtws + (bhC + ct * 16 + ln) * S_ + kb;
      v16bf vB = load16bf(vrow + half * 8, vrow + 16 + half * 8);
      oacc[ct] = wmma_bf16(pA, vB, oacc[ct]);
    }
  }

#pragma unroll
  for (int ct = 0; ct < 4; ++ct)
#pragma unroll
    for (int r = 0; r < 8; ++r) {
      const int qr = q0 + half * 8 + r;
      const int c = ct * 16 + ln;
      const float g = bf2f(gws[(bhS + qr) * C_ + c]);
      const float val = (oacc[ct][r] / lrow[r]) * g;
      aout[((size_t)(b * S_ + qr)) * HD_ + h * C_ + c] = bfbits(val);
    }
}

// ---------------------------------------------------------------------------
// Kernel 3: output projection  out = attn @ wo + bo  (bf16 A, bf16 B^T, f32 out)
// Each wave computes a 16x64 strip (A-tile reused for 4 WMMAs).
// ---------------------------------------------------------------------------
__global__ __launch_bounds__(256) void out_kernel(
    const unsigned short* __restrict__ aws, const unsigned short* __restrict__ woT,
    const float* __restrict__ bo, float* __restrict__ out) {
  const int lane = threadIdx.x & 31;
  const int wave = threadIdx.x >> 5;
  const int ln = lane & 15;
  const int half = lane >> 4;
  const int t = blockIdx.x * 8 + wave;
  const int strip = t & (D_ / 64 - 1);
  const int tm = t >> 4;
  const int row0 = tm * 16, col0 = strip * 64;

  v8f acc[4] = {{}, {}, {}, {}};
  for (int k0 = 0; k0 < HD_; k0 += 32) {
    const unsigned short* ar = aws + (size_t)(row0 + ln) * HD_ + k0;
    v16bf A = load16bf(ar + half * 8, ar + 16 + half * 8);
#pragma unroll
    for (int ct = 0; ct < 4; ++ct) {
      const unsigned short* wrow =
          woT + (size_t)(col0 + ct * 16 + ln) * HD_ + k0;
      v16bf Bt = load16bf(wrow + half * 8, wrow + 16 + half * 8);
      acc[ct] = wmma_bf16(A, Bt, acc[ct]);
    }
  }
#pragma unroll
  for (int ct = 0; ct < 4; ++ct) {
    const int col = col0 + ct * 16 + ln;
    const float bov = bo[col];
#pragma unroll
    for (int r = 0; r < 8; ++r) {
      const int row = row0 + half * 8 + r;
      out[(size_t)row * D_ + col] = acc[ct][r] + bov;
    }
  }
}

extern "C" void kernel_launch(void* const* d_in, const int* in_sizes, int n_in,
                              void* d_out, int out_size, void* d_ws,
                              size_t ws_size, hipStream_t stream) {
  const float* x    = (const float*)d_in[0];
  const float* mask = (const float*)d_in[1];
  const float* bias = (const float*)d_in[2];
  const float* wq   = (const float*)d_in[3];
  const float* wk   = (const float*)d_in[4];
  const float* wv   = (const float*)d_in[5];
  const float* wg   = (const float*)d_in[6];
  const float* bg   = (const float*)d_in[7];
  const float* wo   = (const float*)d_in[8];
  const float* bo   = (const float*)d_in[9];
  float* out = (float*)d_out;

  // Workspace: 5 bf16 activation tensors (8 MB each) + 5 bf16 transposed
  // weights (2 MB each) = 50 MB total.
  const size_t elems = (size_t)B_ * H_ * S_ * C_;   // 4 Mi
  const size_t welems = (size_t)D_ * HD_;           // 1 Mi
  unsigned short* qws  = (unsigned short*)d_ws;
  unsigned short* kws  = qws + elems;
  unsigned short* vtws = kws + elems;
  unsigned short* gws  = vtws + elems;
  unsigned short* aws  = gws + elems;
  unsigned short* tq   = aws + elems;
  unsigned short* tk   = tq + welems;
  unsigned short* tv   = tk + welems;
  unsigned short* tg   = tv + welems;
  unsigned short* toT  = tg + welems;

  wcvt_kernel<<<dim3(D_, 5), dim3(256), 0, stream>>>(wq, wk, wv, wg, wo, tq, tk,
                                                     tv, tg, toT);

  const int strips = (B_ * S_ / 16) * (HD_ / 64);  // 4096 wave-strips
  dim3 pgrid(strips / 8, 1, 4);
  proj_kernel<<<pgrid, dim3(256), 0, stream>>>(x, tq, tk, tv, tg, bg, qws, kws,
                                               vtws, gws);

  dim3 agrid(S_ / 64, H_, B_);  // 4 waves/block, 16 queries/wave
  attn_kernel<<<agrid, dim3(128), 0, stream>>>(qws, kws, vtws, gws, mask, bias,
                                               aws);

  dim3 ogrid(strips / 8, 1, 1);
  out_kernel<<<ogrid, dim3(256), 0, stream>>>(aws, toT, bo, out);
}
